// MultiHeadAttention_41841571398363
// MI455X (gfx1250) — matrix operand order repaired and checked by reading, not compiled, not run on hardware
//
#include <hip/hip_runtime.h>
#include <stdint.h>

#define DEV __device__ __forceinline__

constexpr int S = 2048, E = 1024, D = 64, H = 16;

typedef __bf16 bf16;
typedef bf16     v16bf __attribute__((ext_vector_type(16)));
typedef float    v8f   __attribute__((ext_vector_type(8)));
typedef uint32_t u32x4 __attribute__((ext_vector_type(4)));
typedef int      i32x8 __attribute__((ext_vector_type(8)));
typedef int      i32x4 __attribute__((ext_vector_type(4)));

union ABf { v16bf v; uint32_t u[8]; };

DEV uint16_t f2bf(float f) {                 // native RNE convert (v_cvt_*bf16_f32)
  bf16 h = (bf16)f;
  return __builtin_bit_cast(uint16_t, h);
}

DEV v8f wmma_bf(const ABf& a, const ABf& b, v8f c) {
  return __builtin_amdgcn_wmma_f32_16x16x32_bf16(
      false, a.v, false, b.v, (short)0, c, false, false);
}

// A operand: 16(M) x 32(K) tile from row-major, K-contiguous memory.
DEV void load_a(const uint16_t* base, int ld, int lane, ABf& a) {
  const uint16_t* row = base + (size_t)(lane & 15) * ld;
  const int koff = (lane & 16) ? 8 : 0;
#pragma unroll
  for (int p = 0; p < 8; ++p) {
    int k = (p < 4 ? 2 * p : 8 + 2 * p) + koff;
    a.u[p] = *(const uint32_t*)(row + k);
  }
}

// B operand: 32(K) x 16(N), sourced from row-major W[N][K] (computes X*W^T).
DEV void load_b(const uint16_t* base, int ld, int lane, ABf& b) {
  const uint16_t* row = base + (size_t)(lane & 15) * ld;
  const int koff = (lane & 16) ? 8 : 0;
#pragma unroll
  for (int p = 0; p < 8; ++p)
    b.u[p] = *(const uint32_t*)(row + (p < 4 ? 2 * p : 8 + 2 * p) + koff);
}

// --- Tensor Data Mover: 2D tile (tile_d0 K-contig elems x tile_d1 rows) -> LDS
// D# packed per CDNA5 ISA ch.8 (data_size=1 -> 2-byte elements).
DEV void tdm_load_2d(uint32_t lds_off, const void* gaddr,
                     uint32_t tensor_d0, uint32_t tensor_d1,
                     uint32_t tile_d0, uint32_t tile_d1, uint32_t stride0) {
  uint64_t ga = (uint64_t)gaddr;
  u32x4 g0;
  g0[0] = 1u;                                    // count=1, user descriptor
  g0[1] = lds_off;                               // LDS byte address
  g0[2] = (uint32_t)ga;                          // global_addr[31:0]
  g0[3] = (uint32_t)((ga >> 32) & 0x1FFFFFFu) | (2u << 30);  // addr[56:32] | type=2
  i32x8 g1;
  g1[0] = (int)(1u << 16);                       // data_size=1 (2B), no flags
  g1[1] = (int)((tensor_d0 & 0xFFFFu) << 16);    // tensor_dim0[15:0] -> bits 63:48
  g1[2] = (int)((tensor_d0 >> 16) | ((tensor_d1 & 0xFFFFu) << 16));
  g1[3] = (int)((tensor_d1 >> 16) | (tile_d0 << 16));        // tile_dim0 -> 127:112
  g1[4] = (int)tile_d1;                          // tile_dim1; tile_dim2=0
  g1[5] = (int)stride0;                          // tensor_dim0_stride[31:0]
  g1[6] = 0;
  g1[7] = 0;
  i32x4 z = {0, 0, 0, 0};
#if __clang_major__ >= 23
  i32x8 z8 = {0, 0, 0, 0, 0, 0, 0, 0};
  __builtin_amdgcn_tensor_load_to_lds(g0, g1, z, z, z8, 0);
#else
  __builtin_amdgcn_tensor_load_to_lds(g0, g1, z, z, 0);
#endif
}

// ---------------- stage 1: fp32 -> bf16 conversion ----------------
__global__ void k_cvt(const float* __restrict__ in, uint16_t* __restrict__ out, int n) {
  int i = blockIdx.x * blockDim.x + threadIdx.x;
  int stride = gridDim.x * blockDim.x;
  for (; i < n; i += stride) out[i] = f2bf(in[i]);
}

// ---------------- stage 2: K/Q/V = x @ W^T  ([S,E]x[E,D]) ----------------
__global__ void __launch_bounds__(256, 1)
k_qkv(const uint16_t* __restrict__ xbf,
      const uint16_t* __restrict__ wk,
      const uint16_t* __restrict__ wq,
      const uint16_t* __restrict__ wv,
      uint16_t* __restrict__ ok,
      uint16_t* __restrict__ oq,
      uint16_t* __restrict__ ov) {
  const int lane = threadIdx.x & 31, wave = threadIdx.x >> 5;
  const int mt = blockIdx.x * 8 + wave;               // 0..127 (16-row tile)
  const int which = blockIdx.y;
  const uint16_t* W   = which == 0 ? wk : which == 1 ? wq : wv;
  uint16_t*       out = which == 0 ? ok : which == 1 ? oq : ov;

  v8f acc[4] = {};
  const uint16_t* arow = xbf + (size_t)mt * 16 * E;

  ABf a2[2], b2[2][4];                                // register double buffers
  load_a(arow, E, lane, a2[0]);
#pragma unroll
  for (int t = 0; t < 4; ++t) load_b(W + (size_t)(t * 16) * E, E, lane, b2[0][t]);

  for (int k0 = 0; k0 < E; k0 += 32) {
    const int cur = (k0 >> 5) & 1, nxt = cur ^ 1;
    if (k0 + 32 < E) {                                // prefetch next K-step
      load_a(arow + k0 + 32, E, lane, a2[nxt]);
#pragma unroll
      for (int t = 0; t < 4; ++t)
        load_b(W + (size_t)(t * 16) * E + k0 + 32, E, lane, b2[nxt][t]);
    }
#pragma unroll
    for (int t = 0; t < 4; ++t) acc[t] = wmma_bf(a2[cur], b2[cur][t], acc[t]);
  }
  const int n = lane & 15, hb = (lane & 16) ? 8 : 0;
#pragma unroll
  for (int t = 0; t < 4; ++t)
#pragma unroll
    for (int r = 0; r < 8; ++r) {
      int m = mt * 16 + r + hb;
      out[(size_t)m * D + t * 16 + n] = f2bf(acc[t][r]);
    }
}

// ---------------- stage 3: per-head 64x64 linears (+bias, Q scaled 1/8) ----
__global__ void __launch_bounds__(256, 1)
k_headproj(const uint16_t* __restrict__ kp,
           const uint16_t* __restrict__ qp,
           const uint16_t* __restrict__ vp,
           const uint16_t* __restrict__ hwk,
           const uint16_t* __restrict__ hwq,
           const uint16_t* __restrict__ hwv,
           const float* __restrict__ hbk,
           const float* __restrict__ hbq,
           const float* __restrict__ hbv,
           uint16_t* __restrict__ kh,
           uint16_t* __restrict__ qh,
           uint16_t* __restrict__ vt) {
  const int lane = threadIdx.x & 31, wave = threadIdx.x >> 5;
  const int mt = blockIdx.x * 8 + wave;
  const int h = blockIdx.y;
  const int z = blockIdx.z;                            // 0=K, 1=Q, 2=V
  const uint16_t* X  = z == 0 ? kp  : z == 1 ? qp  : vp;
  const uint16_t* Wh = (z == 0 ? hwk : z == 1 ? hwq : hwv) + (size_t)h * D * D;
  const float*    bh = (z == 0 ? hbk : z == 1 ? hbq : hbv) + (size_t)h * D;

  const uint16_t* arow = X + (size_t)mt * 16 * D;
  ABf a0, a1, b[2][4];                                 // issue all loads up front
  load_a(arow, D, lane, a0);
  load_a(arow + 32, D, lane, a1);
#pragma unroll
  for (int t = 0; t < 4; ++t) {
    load_b(Wh + (size_t)(t * 16) * D, D, lane, b[0][t]);
    load_b(Wh + (size_t)(t * 16) * D + 32, D, lane, b[1][t]);
  }
  v8f acc[4] = {};
#pragma unroll
  for (int t = 0; t < 4; ++t) acc[t] = wmma_bf(a0, b[0][t], acc[t]);
#pragma unroll
  for (int t = 0; t < 4; ++t) acc[t] = wmma_bf(a1, b[1][t], acc[t]);

  const int n = lane & 15, hb = (lane & 16) ? 8 : 0;
  const float qscale = (z == 1) ? 0.125f : 1.0f;       // 1/sqrt(64) folded into Q
#pragma unroll
  for (int t = 0; t < 4; ++t) {
    const float bias = bh[t * 16 + n];
#pragma unroll
    for (int r = 0; r < 8; ++r) {
      float v = (acc[t][r] + bias) * qscale;
      int s = mt * 16 + r + hb;
      int d = t * 16 + n;
      if (z == 2)      vt[(size_t)h * D * S + (size_t)d * S + s] = f2bf(v); // V^T [h][d][s]
      else if (z == 1) qh[(size_t)h * S * D + (size_t)s * D + d] = f2bf(v);
      else             kh[(size_t)h * S * D + (size_t)s * D + d] = f2bf(v);
    }
  }
}

// ---------------- stage 4: flash attention, output into concat layout ------
__global__ void __launch_bounds__(128, 1)
k_attn(const uint16_t* __restrict__ qh,
       const uint16_t* __restrict__ kh,
       const uint16_t* __restrict__ vt,
       uint16_t* __restrict__ cat) {
  __shared__ uint16_t pbuf[4][16 * 32];                // per-wave P-tile transpose buffer
  const int lane = threadIdx.x & 31, wave = threadIdx.x >> 5;
  const int h  = blockIdx.y;
  const int qt = blockIdx.x * 4 + wave;                // 16-query tile, 0..127
  const uint16_t* Q = qh + ((size_t)h * S + (size_t)qt * 16) * D;
  const uint16_t* K = kh + (size_t)h * S * D;
  const uint16_t* V = vt + (size_t)h * D * S;

  ABf qa0, qa1;                                        // Q tile stays in regs
  load_a(Q, D, lane, qa0);
  load_a(Q + 32, D, lane, qa1);

  float mrow[8], lrow[8];
#pragma unroll
  for (int r = 0; r < 8; ++r) { mrow[r] = -3.0e38f; lrow[r] = 0.f; }
  v8f acc[4] = {};
  const int n = lane & 15;

  ABf kb2[2][4];                                       // double-buffered K operands
#pragma unroll
  for (int j = 0; j < 2; ++j) {
    load_b(K + (size_t)(j * 16) * D, D, lane, kb2[0][2 * j]);
    load_b(K + (size_t)(j * 16) * D + 32, D, lane, kb2[0][2 * j + 1]);
  }

  constexpr int NC = S / 32;
  for (int c = 0; c < NC; ++c) {
    const int kb = c * 32, cur = c & 1, nxt = cur ^ 1;
    if (c + 1 < NC) {                                  // prefetch next chunk's K tiles
#pragma unroll
      for (int j = 0; j < 2; ++j) {
        load_b(K + (size_t)(kb + 32 + j * 16) * D, D, lane, kb2[nxt][2 * j]);
        load_b(K + (size_t)(kb + 32 + j * 16) * D + 32, D, lane, kb2[nxt][2 * j + 1]);
      }
    }
    ABf vb[4];                                         // V loads issued before softmax
#pragma unroll
    for (int t = 0; t < 4; ++t)
      load_b(V + (size_t)(t * 16) * S + kb, S, lane, vb[t]);

    v8f sc0 = {}, sc1 = {};
    sc0 = wmma_bf(qa0, kb2[cur][0], sc0);
    sc0 = wmma_bf(qa1, kb2[cur][1], sc0);
    sc1 = wmma_bf(qa0, kb2[cur][2], sc1);
    sc1 = wmma_bf(qa1, kb2[cur][3], sc1);

    uint16_t* pb = pbuf[wave];
#pragma unroll
    for (int r = 0; r < 8; ++r) {                      // online softmax (hides V latency)
      float mx = fmaxf(sc0[r], sc1[r]);
#pragma unroll
      for (int off = 1; off < 16; off <<= 1) mx = fmaxf(mx, __shfl_xor(mx, off, 32));
      float mnew  = fmaxf(mrow[r], mx);
      float alpha = __expf(mrow[r] - mnew);
      mrow[r] = mnew;
      float p0 = __expf(sc0[r] - mnew);
      float p1 = __expf(sc1[r] - mnew);
      float ps = p0 + p1;
#pragma unroll
      for (int off = 1; off < 16; off <<= 1) ps += __shfl_xor(ps, off, 32);
      lrow[r] = lrow[r] * alpha + ps;
#pragma unroll
      for (int t = 0; t < 4; ++t) acc[t][r] *= alpha;
      int m = r + ((lane & 16) ? 8 : 0);
      pb[m * 32 + n]      = f2bf(p0);
      pb[m * 32 + 16 + n] = f2bf(p1);
    }
    __builtin_amdgcn_wave_barrier();                   // keep DS store->load ordered
    ABf pa; load_a(pb, 32, lane, pa);                  // P as A operand [16q x 32keys]
#pragma unroll
    for (int t = 0; t < 4; ++t) acc[t] = wmma_bf(pa, vb[t], acc[t]);
  }
  const int hb = (lane & 16) ? 8 : 0;
#pragma unroll
  for (int t = 0; t < 4; ++t)
#pragma unroll
    for (int r = 0; r < 8; ++r) {
      int s = qt * 16 + r + hb;
      cat[(size_t)s * E + h * D + t * 16 + n] = f2bf(acc[t][r] / lrow[r]);
    }
}

// ---------------- stage 5: out = cat @ Wo^T + bo, +x, LayerNorm ------------
// Wo streamed through LDS by the Tensor Data Mover, double buffered.
__global__ void __launch_bounds__(256, 1)
k_out(const uint16_t* __restrict__ cat,
      const uint16_t* __restrict__ wo,
      const float* __restrict__ bo,
      const float* __restrict__ x,
      const float* __restrict__ gamma,
      const float* __restrict__ beta,
      float* __restrict__ out) {
  constexpr int KC = 32;                  // K elems per chunk
  constexpr int NCH = E / KC;             // 32 chunks
  extern __shared__ uint16_t wdyn[];      // 2 x [E rows][KC] bf16 = 2 x 64 KB
  __shared__ float red[8][16][2];

  const int lane = threadIdx.x & 31, wave = threadIdx.x >> 5;
  const int mt = blockIdx.x;              // 16-row strip
  const int cbase = wave * 128;           // this wave's 128 output columns

  if (wave == 0) {                        // TDM prologue: chunks 0 and 1 in flight
    tdm_load_2d((uint32_t)(size_t)(void*)&wdyn[0], wo, E, E, KC, E, E);
    tdm_load_2d((uint32_t)(size_t)(void*)&wdyn[(size_t)E * KC], wo + KC, E, E, KC, E, E);
  }

  v8f acc[8] = {};
  const uint16_t* arow = cat + (size_t)mt * 16 * E;
  ABf a2[2];
  load_a(arow, E, lane, a2[0]);

  for (int c = 0; c < NCH; ++c) {
    if (wave == 0) {
      if (c + 1 < NCH) __builtin_amdgcn_s_wait_tensorcnt(1);
      else             __builtin_amdgcn_s_wait_tensorcnt(0);
    }
    __syncthreads();                      // chunk c resident in buf[c&1]
    if (c + 1 < NCH) load_a(arow + (c + 1) * KC, E, lane, a2[(c + 1) & 1]);
    const uint16_t* wb = wdyn + (size_t)(c & 1) * E * KC;
#pragma unroll
    for (int t = 0; t < 8; ++t) {
      ABf b; load_b(wb + (size_t)(cbase + t * 16) * KC, KC, lane, b);
      acc[t] = wmma_bf(a2[c & 1], b, acc[t]);
    }
    __syncthreads();                      // everyone done reading buf[c&1]
    if (wave == 0 && c + 2 < NCH)
      tdm_load_2d((uint32_t)(size_t)(void*)&wdyn[(size_t)(c & 1) * E * KC],
                  wo + (size_t)(c + 2) * KC, E, E, KC, E, E);
  }

  const int n = lane & 15, hb = (lane & 16) ? 8 : 0;
#pragma unroll
  for (int t = 0; t < 8; ++t) {
    int e = cbase + t * 16 + n;
    float bias = bo[e];
#pragma unroll
    for (int r = 0; r < 8; ++r) {
      int s = mt * 16 + r + hb;
      acc[t][r] += bias + x[(size_t)s * E + e];        // bias + residual
    }
  }
#pragma unroll
  for (int r = 0; r < 8; ++r) {                        // per-wave row partials
    float sm = 0.f, sq = 0.f;
#pragma unroll
    for (int t = 0; t < 8; ++t) { float v = acc[t][r]; sm += v; sq += v * v; }
#pragma unroll
    for (int off = 1; off < 16; off <<= 1) {
      sm += __shfl_xor(sm, off, 32);
      sq += __shfl_xor(sq, off, 32);
    }
    if (n == 0) { red[wave][r + hb][0] = sm; red[wave][r + hb][1] = sq; }
  }
  __syncthreads();
#pragma unroll
  for (int r = 0; r < 8; ++r) {
    int row = r + hb;
    float sm = 0.f, sq = 0.f;
#pragma unroll
    for (int w2 = 0; w2 < 8; ++w2) { sm += red[w2][row][0]; sq += red[w2][row][1]; }
    float mu   = sm * (1.0f / E);
    float var  = sq * (1.0f / E) - mu * mu;
    float rstd = rsqrtf(var + 1e-5f);
    int s = mt * 16 + row;
#pragma unroll
    for (int t = 0; t < 8; ++t) {
      int e = cbase + t * 16 + n;
      out[(size_t)s * E + e] = (acc[t][r] - mu) * rstd * gamma[e] + beta[e];
    }
  }
}

extern "C" void kernel_launch(void* const* d_in, const int* in_sizes, int n_in,
                              void* d_out, int out_size, void* d_ws, size_t ws_size,
                              hipStream_t stream) {
  (void)in_sizes; (void)n_in; (void)out_size; (void)ws_size;
  const float* x     = (const float*)d_in[0];
  const float* Wk    = (const float*)d_in[1];
  const float* Wq    = (const float*)d_in[2];
  const float* Wv    = (const float*)d_in[3];
  const float* hWk   = (const float*)d_in[4];
  const float* hbk   = (const float*)d_in[5];
  const float* hWv   = (const float*)d_in[6];
  const float* hbv   = (const float*)d_in[7];
  const float* hWq   = (const float*)d_in[8];
  const float* hbq   = (const float*)d_in[9];
  const float* Wo    = (const float*)d_in[10];
  const float* bo    = (const float*)d_in[11];
  const float* gamma = (const float*)d_in[12];
  const float* beta  = (const float*)d_in[13];

  uint16_t* w = (uint16_t*)d_ws;
  constexpr size_t XBF = 0;
  constexpr size_t WKB = XBF + (size_t)S * E;
  constexpr size_t WQB = WKB + (size_t)D * E;
  constexpr size_t WVB = WQB + (size_t)D * E;
  constexpr size_t HWK = WVB + (size_t)D * E;
  constexpr size_t HWQ = HWK + (size_t)H * D * D;
  constexpr size_t HWV = HWQ + (size_t)H * D * D;
  constexpr size_t WOB = HWV + (size_t)H * D * D;
  constexpr size_t KP  = WOB + (size_t)E * E;
  constexpr size_t QP  = KP + (size_t)S * D;
  constexpr size_t VP  = QP + (size_t)S * D;
  constexpr size_t KH  = VP + (size_t)S * D;
  constexpr size_t QH  = KH + (size_t)H * S * D;
  constexpr size_t VT  = QH + (size_t)H * S * D;
  constexpr size_t CAT = VT + (size_t)H * S * D;

  auto cvt = [&](const float* src, size_t off, int nElem) {
    int blocks = (nElem + 255) / 256;
    if (blocks > 2048) blocks = 2048;
    k_cvt<<<blocks, 256, 0, stream>>>(src, w + off, nElem);
  };
  cvt(x,   XBF, S * E);
  cvt(Wk,  WKB, D * E);
  cvt(Wq,  WQB, D * E);
  cvt(Wv,  WVB, D * E);
  cvt(hWk, HWK, H * D * D);
  cvt(hWq, HWQ, H * D * D);
  cvt(hWv, HWV, H * D * D);
  cvt(Wo,  WOB, E * E);

  k_qkv<<<dim3(S / 16 / 8, 3), 256, 0, stream>>>(
      w + XBF, w + WKB, w + WQB, w + WVB, w + KP, w + QP, w + VP);

  k_headproj<<<dim3(S / 16 / 8, H, 3), 256, 0, stream>>>(
      w + KP, w + QP, w + VP, w + HWK, w + HWQ, w + HWV,
      hbk, hbq, hbv, w + KH, w + QH, w + VT);

  k_attn<<<dim3(S / 16 / 4, H), 128, 0, stream>>>(
      w + QH, w + KH, w + VT, w + CAT);

  k_out<<<dim3(S / 16), 256, 2u * E * 32 * sizeof(uint16_t), stream>>>(
      w + CAT, w + WOB, bo, x, gamma, beta, (float*)d_out);
}
